// SelfLocalReferenceAttn_86801289052454
// MI455X (gfx1250) — matrix operand order repaired and checked by reading, not compiled, not run on hardware
//
#include <hip/hip_runtime.h>

typedef __attribute__((ext_vector_type(16))) _Float16 v16h;
typedef __attribute__((ext_vector_type(8)))  float    v8f;
typedef __attribute__((ext_vector_type(4)))  unsigned int u32x4;
typedef __attribute__((ext_vector_type(8)))  int      i32x8;
typedef __attribute__((ext_vector_type(4)))  int      i32x4;

#define S_LEN  1024
#define BATCH  8
#define DMODEL 512
#define NHEAD  8
#define HDIM   64
#define DFF    256
#define NLAYER 2
#define RADIUS 20
#define NTOK   (S_LEN*BATCH)   // 8192 tokens

#define BM 128
#define BN 64
#define BK 32

#if defined(__has_builtin)
#if __has_builtin(__builtin_amdgcn_tensor_load_to_lds)
#define HAVE_TDM 1
#endif
#endif
#ifndef HAVE_TDM
#define HAVE_TDM 0
#endif

// ---- WMMA fragment loaders (wave32 layouts per CDNA5 ISA 7.12.2) ----
// A fragment: 16x32 f16 tile in LDS, row stride BK.
// lanes 0-15: M=lane, K={0..7,16..23}; lanes 16-31: M=lane-16, K={8..15,24..31}
__device__ __forceinline__ v16h frag_A_lds(const _Float16* base) {
  const int lane = threadIdx.x & 31;
  const int m    = lane & 15;
  const int kb   = (lane < 16) ? 0 : 8;
  const _Float16* p = base + m * BK;
  v16h a;
#pragma unroll
  for (int j = 0; j < 8; ++j) {
    const int k = kb + ((j < 4) ? (2*j) : (16 + 2*(j-4)));
    a[2*j]   = p[k];
    a[2*j+1] = p[k+1];
  }
  return a;
}
// B fragment: 32x16 (KxN), staged as W rows (N-major, K contiguous).
// lanes 0-15: N=lane, K={0..7,16..23}; lanes 16-31: N=lane-16, K={8..15,24..31}
__device__ __forceinline__ v16h frag_B_lds(const _Float16* base) {
  const int lane = threadIdx.x & 31;
  const int n    = lane & 15;
  const int koff = (lane < 16) ? 0 : 8;
  const _Float16* p = base + n * BK + koff;
  v16h b;
#pragma unroll
  for (int j = 0; j < 8; ++j) { const int kk = (j < 4) ? (2*j) : (16 + 2*(j-4)); b[2*j] = p[kk]; b[2*j+1] = p[kk+1]; }
  return b;
}

// ---- C[M,N] = A[M,K] @ W[N,K]^T + bias[N]; A/W are f16, C is f32 ----
// 128x64 block tile, double-buffered LDS; 8 waves = 4(m) x 2(n), wave -> 32x32.
// Weight tile fetched by the Tensor Data Mover when available.
__global__ __launch_bounds__(256) void gemm_f16_wmma(
    const _Float16* __restrict__ A, const _Float16* __restrict__ W,
    const float* __restrict__ bias, float* __restrict__ C,
    int M, int N, int K)
{
  __shared__ _Float16 As[2][BM][BK];   // 2 x 8 KB
  __shared__ _Float16 Bs[2][BN][BK];   // 2 x 4 KB
  const int tiles_n = N / BN;
  const int bm = blockIdx.x / tiles_n;
  const int bn = blockIdx.x % tiles_n;
  const int row0 = bm * BM, col0 = bn * BN;
  const int tid  = threadIdx.x;
  const int wave = tid >> 5;
  const int lane = tid & 31;
  const int wm   = wave & 3;    // 0..3 -> 32-row sub-tile
  const int wn   = wave >> 2;   // 0..1 -> 32-col sub-tile

  // cooperative A load: thread -> row ar, 16 halves at col ac (2 x b128)
  const int ar = tid >> 1, ac = (tid & 1) * 16;
  const _Float16* gA = A + (size_t)(row0 + ar) * K + ac;
  // cooperative B fallback: thread -> row br, 8 halves at col bc (1 x b128)
  const int br = tid >> 2, bc = (tid & 3) * 8;
  const _Float16* gB = W + (size_t)(col0 + br) * K + bc;

  const int nsteps = K / BK;

  auto loadA = [&](int buf, int k0) {
    const uint4* p = (const uint4*)(gA + k0);
    uint4 v0 = p[0], v1 = p[1];
    if (k0 + BK < K) __builtin_prefetch(gA + k0 + BK, 0, 1);
    *(uint4*)&As[buf][ar][ac]     = v0;
    *(uint4*)&As[buf][ar][ac + 8] = v1;
  };
#if HAVE_TDM
  auto loadB = [&](int buf, int k0) {
    if (wave == 0) {
      const unsigned lds_lo = (unsigned)(uintptr_t)&Bs[buf][0][0];
      const unsigned long long ga =
          (unsigned long long)(uintptr_t)(W + (size_t)col0 * K + k0);
      u32x4 g0 = {};
      g0[0] = 1u;                                   // count=1, user mode
      g0[1] = lds_lo;                               // lds_addr
      g0[2] = (unsigned)(ga & 0xffffffffu);         // global_addr[31:0]
      g0[3] = (unsigned)((ga >> 32) & 0x01ffffffu)  // global_addr[56:32]
              | 0x80000000u;                        // type=2 (image)
      i32x8 g1 = {};
      g1[0] = 1 << 16;                              // data_size=1 -> 2 bytes
      g1[1] = (int)((K & 0xffff) << 16);            // tensor_dim0[15:0]
      g1[2] = (int)(((unsigned)K >> 16) | (((unsigned)N & 0xffff) << 16));
      g1[3] = (int)((((unsigned)N >> 16) & 0xffffu) | ((unsigned)BK << 16));
      g1[4] = BN;                                   // tile_dim1=64, tile_dim2=0
      g1[5] = (int)K;                               // tensor_dim0_stride[31:0]
      g1[6] = 0;
      g1[7] = 0;
      i32x4 g2 = {}, g3 = {};                       // 2D tensor: groups 2/3 zero
#if __clang_major__ >= 23
      i32x8 g4 = {};
      __builtin_amdgcn_tensor_load_to_lds(g0, g1, g2, g3, g4, 0);
#else
      __builtin_amdgcn_tensor_load_to_lds(g0, g1, g2, g3, 0);
#endif
    }
  };
#else
  auto loadB = [&](int buf, int k0) {
    const uint4* p = (const uint4*)(gB + k0);
    *(uint4*)&Bs[buf][br][bc] = p[0];
  };
#endif

  // prologue: fill buffer 0
  loadA(0, 0);
  loadB(0, 0);
#if HAVE_TDM
  if (wave == 0) __builtin_amdgcn_s_wait_tensorcnt(0);
#endif
  __syncthreads();

  v8f acc00 = {}, acc01 = {}, acc10 = {}, acc11 = {};
  for (int i = 0; i < nsteps; ++i) {
    const int buf = i & 1;
    if (i + 1 < nsteps) {          // stage next slab while computing this one
      loadA(buf ^ 1, (i + 1) * BK);
      loadB(buf ^ 1, (i + 1) * BK);
    }
    const v16h a0 = frag_A_lds(&As[buf][wm*32     ][0]);
    const v16h a1 = frag_A_lds(&As[buf][wm*32 + 16][0]);
    const v16h b0 = frag_B_lds(&Bs[buf][wn*32     ][0]);
    const v16h b1 = frag_B_lds(&Bs[buf][wn*32 + 16][0]);
    acc00 = __builtin_amdgcn_wmma_f32_16x16x32_f16(false, a0, false, b0,
                                                   (short)0, acc00, false, false);
    acc01 = __builtin_amdgcn_wmma_f32_16x16x32_f16(false, a0, false, b1,
                                                   (short)0, acc01, false, false);
    acc10 = __builtin_amdgcn_wmma_f32_16x16x32_f16(false, a1, false, b0,
                                                   (short)0, acc10, false, false);
    acc11 = __builtin_amdgcn_wmma_f32_16x16x32_f16(false, a1, false, b1,
                                                   (short)0, acc11, false, false);
#if HAVE_TDM
    if (wave == 0 && i + 1 < nsteps) __builtin_amdgcn_s_wait_tensorcnt(0);
#endif
    __syncthreads();
  }

  // C/D layout: col n = lane&15 (+16 for second tile), rows m = (lane<16?0:8)+j
  const int n  = lane & 15;
  const int mo = (lane < 16) ? 0 : 8;
  const int gr = row0 + wm*32 + mo;
  const int gc = col0 + wn*32 + n;
  const float bv0 = bias[gc];
  const float bv1 = bias[gc + 16];
#pragma unroll
  for (int j = 0; j < 8; ++j) {
    C[(size_t)(gr + j)      * N + gc]      = acc00[j] + bv0;
    C[(size_t)(gr + j)      * N + gc + 16] = acc01[j] + bv1;
    C[(size_t)(gr + 16 + j) * N + gc]      = acc10[j] + bv0;
    C[(size_t)(gr + 16 + j) * N + gc + 16] = acc11[j] + bv1;
  }
}

// ---- banded flash attention: one wave per (query, batch, head) ----
// qkv f32 [NTOK, 3*DMODEL], row = s*B + b; writes f16 output for the WMMA proj.
__global__ __launch_bounds__(256) void attn_band(
    const float* __restrict__ qkv, _Float16* __restrict__ out16)
{
  const int gwave = (int)((blockIdx.x * 256 + threadIdx.x) >> 5);
  const int lane  = threadIdx.x & 31;
  const int h  = gwave & (NHEAD - 1);
  const int b  = (gwave >> 3) & (BATCH - 1);
  const int qi = gwave >> 6;
  const int d0 = lane * 2;
  const size_t qrow = (size_t)(qi * BATCH + b) * (3 * DMODEL);
  const float scale = 0.125f;    // 1/sqrt(64)
  const float q0 = qkv[qrow + h*HDIM + d0]     * scale;
  const float q1 = qkv[qrow + h*HDIM + d0 + 1] * scale;
  int jlo = qi - (RADIUS - 1); if (jlo < 0) jlo = 0;
  int jhi = qi + (RADIUS - 1); if (jhi > S_LEN - 1) jhi = S_LEN - 1;
  float m = -3.0e38f, l = 0.0f, acc0 = 0.0f, acc1 = 0.0f;
  for (int j = jlo; j <= jhi; ++j) {
    const size_t krow = (size_t)(j * BATCH + b) * (3 * DMODEL);
    const float k0 = qkv[krow + DMODEL + h*HDIM + d0];
    const float k1 = qkv[krow + DMODEL + h*HDIM + d0 + 1];
    float p = q0 * k0 + q1 * k1;
#pragma unroll
    for (int off = 16; off; off >>= 1) p += __shfl_xor(p, off, 32);
    const float v0 = qkv[krow + 2*DMODEL + h*HDIM + d0];
    const float v1 = qkv[krow + 2*DMODEL + h*HDIM + d0 + 1];
    const float mn  = fmaxf(m, p);
    const float cor = __expf(m - mn);
    const float w   = __expf(p - mn);
    l    = l    * cor + w;
    acc0 = acc0 * cor + w * v0;
    acc1 = acc1 * cor + w * v1;
    m = mn;
  }
  const float inv = 1.0f / l;
  _Float16* op = out16 + (size_t)(qi * BATCH + b) * DMODEL + h*HDIM + d0;
  op[0] = (_Float16)(acc0 * inv);
  op[1] = (_Float16)(acc1 * inv);
}

// ---- LayerNorm rows of 512; optional residual; optional f16 shadow output ----
__global__ __launch_bounds__(256) void ln_row(
    const float* __restrict__ inp, const float* __restrict__ res,
    const float* __restrict__ g, const float* __restrict__ be,
    float* __restrict__ out, _Float16* __restrict__ out16, int has_res)
{
  const int row = blockIdx.x;
  const int t   = threadIdx.x;
  const float* pin = inp + (size_t)row * DMODEL;
  float v0 = pin[t], v1 = pin[t + 256];
  if (has_res) {
    const float* pr = res + (size_t)row * DMODEL;
    v0 += pr[t]; v1 += pr[t + 256];
  }
  float s  = v0 + v1;
  float sq = v0 * v0 + v1 * v1;
#pragma unroll
  for (int off = 16; off; off >>= 1) {
    s  += __shfl_xor(s,  off, 32);
    sq += __shfl_xor(sq, off, 32);
  }
  __shared__ float ss[8], ssq[8];
  const int wv = t >> 5, ln = t & 31;
  if (ln == 0) { ss[wv] = s; ssq[wv] = sq; }
  __syncthreads();
  float S = 0.0f, Q = 0.0f;
#pragma unroll
  for (int i = 0; i < 8; ++i) { S += ss[i]; Q += ssq[i]; }
  const float mu   = S * (1.0f / DMODEL);
  const float var  = Q * (1.0f / DMODEL) - mu * mu;
  const float rstd = rsqrtf(var + 1e-5f);
  const float y0 = (v0 - mu) * rstd * g[t]       + be[t];
  const float y1 = (v1 - mu) * rstd * g[t + 256] + be[t + 256];
  float* po = out + (size_t)row * DMODEL;
  po[t]       = y0;
  po[t + 256] = y1;
  if (out16) {
    _Float16* ph = out16 + (size_t)row * DMODEL;
    ph[t]       = (_Float16)y0;
    ph[t + 256] = (_Float16)y1;
  }
}

// ---- exact GELU: f32 in -> f16 out (feeds WMMA FFN2) ----
__global__ __launch_bounds__(256) void gelu_f16(
    const float* __restrict__ x, _Float16* __restrict__ y, int n)
{
  const int i = blockIdx.x * 256 + threadIdx.x;
  if (i < n) {
    const float v = x[i];
    y[i] = (_Float16)(0.5f * v * (1.0f + erff(v * 0.70710678118654752f)));
  }
}

// ---- f32 -> f16 pack, 4 elements/thread (n multiple of 1024) ----
__global__ __launch_bounds__(256) void f32_to_f16_vec(
    const float* __restrict__ in, _Float16* __restrict__ out, int n)
{
  const int i = (blockIdx.x * 256 + threadIdx.x) * 4;
  if (i < n) {
    const float4 v = *(const float4*)(in + i);
    out[i]     = (_Float16)v.x;
    out[i + 1] = (_Float16)v.y;
    out[i + 2] = (_Float16)v.z;
    out[i + 3] = (_Float16)v.w;
  }
}

extern "C" void kernel_launch(void* const* d_in, const int* in_sizes, int n_in,
                              void* d_out, int out_size, void* d_ws, size_t ws_size,
                              hipStream_t stream) {
  (void)in_sizes; (void)n_in; (void)out_size; (void)ws_size;
  const float* x_in = (const float*)d_in[0];
  float* ws = (float*)d_ws;

  // f32 region
  float* buf_x   = ws;                                   // NTOK*512 master act
  float* buf_big = buf_x + (size_t)NTOK * DMODEL;        // NTOK*1536 qkv
  float* buf_tmp = buf_big;                              // reuse: NTOK*512
  float* buf_ff  = buf_big + (size_t)NTOK * DMODEL;      // reuse: NTOK*256
  // f16 region
  _Float16* hp = (_Float16*)(buf_big + (size_t)NTOK * 3 * DMODEL);
  _Float16* x16    = hp;  hp += (size_t)NTOK * DMODEL;   // GEMM input (act)
  _Float16* attn16 = hp;  hp += (size_t)NTOK * DMODEL;   // attention out
  _Float16* ff16   = hp;  hp += (size_t)NTOK * DFF;      // gelu out
  _Float16* w16[NLAYER][4];
  const int wsz[4] = {3*DMODEL*DMODEL, DMODEL*DMODEL, DFF*DMODEL, DMODEL*DFF};
  for (int L = 0; L < NLAYER; ++L)
    for (int w = 0; w < 4; ++w) { w16[L][w] = hp; hp += wsz[w]; }

  // one-time (per launch) f16 packing: weights + input activations
  f32_to_f16_vec<<<dim3(NTOK*DMODEL/1024), 256, 0, stream>>>(x_in, x16, NTOK*DMODEL);
  for (int L = 0; L < NLAYER; ++L) {
    const int widx[4] = {0, 2, 6, 8};   // Wqkv, Wo, W1, W2 within layer params
    for (int w = 0; w < 4; ++w) {
      const float* src = (const float*)d_in[1 + L * 14 + widx[w]];
      f32_to_f16_vec<<<dim3(wsz[w]/1024), 256, 0, stream>>>(src, w16[L][w], wsz[w]);
    }
  }

  for (int layer = 0; layer < NLAYER; ++layer) {
    const float* P[14];
    for (int i = 0; i < 14; ++i) P[i] = (const float*)d_in[1 + layer * 14 + i];
    const float *bqkv = P[1], *bo = P[3];
    const float *ln1g = P[4], *ln1b = P[5];
    const float *b1 = P[7], *b2 = P[9];
    const float *ln2g = P[10], *ln2b = P[11], *lnfg = P[12], *lnfb = P[13];
    const float* xin_f32 = (layer == 0) ? x_in : buf_x;

    // 1) qkv = x @ Wqkv^T + bqkv            [8192 x 1536] f32
    gemm_f16_wmma<<<dim3((NTOK/BM) * ((3*DMODEL)/BN)), 256, 0, stream>>>(
        x16, w16[layer][0], bqkv, buf_big, NTOK, 3*DMODEL, DMODEL);
    // 2) banded attention -> f16            [8192 x 512]
    attn_band<<<dim3(NTOK * NHEAD / 8), 256, 0, stream>>>(buf_big, attn16);
    // 3) out projection                     [8192 x 512] f32
    gemm_f16_wmma<<<dim3((NTOK/BM) * (DMODEL/BN)), 256, 0, stream>>>(
        attn16, w16[layer][1], bo, buf_tmp, NTOK, DMODEL, DMODEL);
    // 4) x = LN(x + proj)  (also emit f16 for FFN1)
    ln_row<<<dim3(NTOK), 256, 0, stream>>>(buf_tmp, xin_f32, ln1g, ln1b,
                                           buf_x, x16, 1);
    // 5) h = x @ W1^T + b1                  [8192 x 256] f32
    gemm_f16_wmma<<<dim3((NTOK/BM) * (DFF/BN)), 256, 0, stream>>>(
        x16, w16[layer][2], b1, buf_ff, NTOK, DFF, DMODEL);
    // 6) h16 = gelu(h)
    gelu_f16<<<dim3(NTOK * DFF / 256), 256, 0, stream>>>(buf_ff, ff16, NTOK * DFF);
    // 7) f = h16 @ W2^T + b2                [8192 x 512] f32
    gemm_f16_wmma<<<dim3((NTOK/BM) * (DMODEL/BN)), 256, 0, stream>>>(
        ff16, w16[layer][3], b2, buf_tmp, NTOK, DMODEL, DFF);
    // 8) x = LN(x + f)
    ln_row<<<dim3(NTOK), 256, 0, stream>>>(buf_tmp, buf_x, ln2g, ln2b,
                                           buf_x, nullptr, 1);
    // 9) x = LN(x) per-block final norm (f16 shadow feeds next layer's QKV)
    ln_row<<<dim3(NTOK), 256, 0, stream>>>(buf_x, nullptr, lnfg, lnfb,
                                           buf_x, x16, 0);
  }
  hipMemcpyAsync(d_out, buf_x, (size_t)NTOK * DMODEL * sizeof(float),
                 hipMemcpyDeviceToDevice, stream);
}
